// XposMultiheadAttention_35210141892844
// MI455X (gfx1250) — matrix operand order repaired and checked by reading, not compiled, not run on hardware
//
#include <hip/hip_runtime.h>
#include <hip/hip_bf16.h>

// ---------------------------------------------------------------------------
// XPos multihead attention for MI455X (gfx1250), wave32 + WMMA bf16.
//   B=2, T=S=2048, E=1024, H=16, D=64.
// All GEMMs use v_wmma_f32_16x16x32_bf16 (f32 accumulate). Attention is
// flash-style with K/V tiles staged to LDS via async global->LDS copies
// (ASYNCcnt), double-buffered to overlap DMA with WMMA. Workspace (64 MB):
//   Xq,Xk,Xv bf16 [4096][1024]   (3 x 8MB)
//   Wq,Wk,Wv,Wo bf16 [1024][1024](4 x 2MB)
//   Qb,Kb bf16 [B*H][2048][64], Vt bf16 [B*H][64][2048], At bf16 [B*T][1024]
// ---------------------------------------------------------------------------

typedef __attribute__((ext_vector_type(16))) __bf16 v16bf;
typedef __attribute__((ext_vector_type(8)))  float  v8f;

#define TPOS 2048
#define EDIM 1024
#define HDIM 64
#define NHEAD 16
#define NROWS 4096  /* B*T */
#define LDP 72      /* padded LDS row stride in halves (144B = 36 banks) */

__device__ inline unsigned short f2bf(float f) {
  unsigned int b = __float_as_uint(f);
  unsigned int r = b + 0x7FFFu + ((b >> 16) & 1u);   // round-to-nearest-even
  return (unsigned short)(r >> 16);
}

__device__ inline v8f wmma_bf16(v16bf a, v16bf b, v8f c) {
  return __builtin_amdgcn_wmma_f32_16x16x32_bf16(false, a, false, b,
                                                 (short)0, c, false, false);
}

// A fragment, 16x32 bf16 (ISA 7.12.2): lane row = lane&15;
// lanes 0-15 hold K = {kBase+0..7, kBase+16..23}, lanes 16-31 K = {+8..15, +24..31}.
__device__ inline v16bf load_a_frag(const unsigned short* __restrict__ X, int ld,
                                    int rowBase, int kBase, int lane) {
  const int row = rowBase + (lane & 15);
  const int kh  = (lane < 16) ? 0 : 8;
  const unsigned short* p = X + (size_t)row * ld + kBase + kh;
  union { v16bf v; uint4 q[2]; } u;
  u.q[0] = *(const uint4*)(p);        // K = kBase+kh .. +7
  u.q[1] = *(const uint4*)(p + 16);   // K = kBase+kh+16 .. +23
  return u.v;
}

// B fragment, 32x16 bf16: lane col N = lane&15;
// lanes 0-15 hold K = {kBase+0..7, kBase+16..23}, lanes 16-31 K = {+8..15, +24..31}.
__device__ inline v16bf load_b_frag(const unsigned short* __restrict__ W, int ld,
                                    int colBase, int kBase, int lane) {
  const int n  = colBase + (lane & 15);
  const int ks = kBase + ((lane < 16) ? 0 : 8);
  const unsigned short* p = W + (size_t)n * ld + ks;
  union { v16bf v; uint4 q[2]; } u;
  u.q[0] = *(const uint4*)(p);
  u.q[1] = *(const uint4*)(p + 16);
  return u.v;
}

// Async 16B global -> LDS copy (GLOBAL_LOAD_ASYNC_TO_LDS_B128, ASYNCcnt).
// vdst VGPR carries the LDS byte address (low 32 bits of generic pointer).
__device__ inline void async_ld16(unsigned lds_addr, unsigned long long gaddr) {
  asm volatile("global_load_async_to_lds_b128 %0, %1, off"
               :: "v"(lds_addr), "v"(gaddr) : "memory");
}

// ---------------------------------------------------------------------------
// f32 -> bf16 bulk convert (vectorized x4)
// ---------------------------------------------------------------------------
__global__ void conv_kernel(const float* __restrict__ src,
                            unsigned short* __restrict__ dst, int n4) {
  int i = blockIdx.x * blockDim.x + threadIdx.x;
  if (i < n4) {
    float4 f = ((const float4*)src)[i];
    ushort4 u;
    u.x = f2bf(f.x); u.y = f2bf(f.y); u.z = f2bf(f.z); u.w = f2bf(f.w);
    ((ushort4*)dst)[i] = u;
  }
}

// ---------------------------------------------------------------------------
// GEMM  Y[row, col] = X[row,:] . W[col,:] + bias[col], fused epilogue.
// MODE 0: q  -> *0.125, xpos(up),  store bf16 [b][h][t][d]
// MODE 1: k  ->          xpos(down),store bf16 [b][h][t][d]
// MODE 2: v  ->                     store bf16 [b][h][d][t]  (transposed)
// MODE 3: o  ->                     store f32  [b][t][e]     (= d_out)
// Block: 256 thr = 8 waves; tile 128(M) x 64(N); wave tile 32x32 (2x2 WMMA).
// ---------------------------------------------------------------------------
template <int MODE>
__global__ __launch_bounds__(256) void gemm_kernel(
    const unsigned short* __restrict__ X, const unsigned short* __restrict__ W,
    const float* __restrict__ bias, unsigned short* __restrict__ outb,
    float* __restrict__ outf) {
  const int tid = threadIdx.x;
  const int lane = tid & 31, wave = tid >> 5;
  const int wm = wave & 3, wn = wave >> 2;
  const int r0 = blockIdx.y * 128 + wm * 32;
  const int c0 = blockIdx.x * 64 + wn * 32;

  v8f acc[2][2];
#pragma unroll
  for (int i = 0; i < 2; ++i)
#pragma unroll
    for (int j = 0; j < 2; ++j) acc[i][j] = (v8f){0,0,0,0,0,0,0,0};

  for (int k0 = 0; k0 < EDIM; k0 += 32) {
    v16bf a0 = load_a_frag(X, EDIM, r0,      k0, lane);
    v16bf a1 = load_a_frag(X, EDIM, r0 + 16, k0, lane);
    v16bf b0 = load_b_frag(W, EDIM, c0,      k0, lane);
    v16bf b1 = load_b_frag(W, EDIM, c0 + 16, k0, lane);
    acc[0][0] = wmma_bf16(a0, b0, acc[0][0]);
    acc[0][1] = wmma_bf16(a0, b1, acc[0][1]);
    acc[1][0] = wmma_bf16(a1, b0, acc[1][0]);
    acc[1][1] = wmma_bf16(a1, b1, acc[1][1]);
  }

  // Epilogue. C layout: VGPR r holds M = 8*(lane>=16)+r, N = lane&15.
#pragma unroll
  for (int mt = 0; mt < 2; ++mt) {
#pragma unroll
    for (int nt = 0; nt < 2; ++nt) {
      const int col = c0 + nt * 16 + (lane & 15);
      const float bcol = bias[col];
      float freq = 0.f, lbase = 0.f;
      if (MODE == 0 || MODE == 1) {
        const float fi = (float)((col & 63) >> 1);
        freq  = __expf(fi * -0.28782313662425572f);        // 10000^(-i/32)
        lbase = __logf((2.0f * fi + 25.6f) * (1.0f / 89.6f));
      }
#pragma unroll
      for (int r = 0; r < 8; ++r) {
        const int row = r0 + mt * 16 + ((lane >> 4) << 3) + r;
        float val = acc[mt][nt][r] + bcol;
        if (MODE == 0) val *= 0.125f;                      // headdim^-0.5
        if (MODE == 0 || MODE == 1) {
          const float prt = __shfl_xor(val, 1, 32);        // rotary partner
          const float ft  = (float)(row & (TPOS - 1));
          const float ang = ft * freq;
          const float pw  = (ft - 1024.0f) * (1.0f / 1024.0f);
          const float sc  = __expf(lbase * ((MODE == 1) ? -pw : pw));
          const float cs  = __cosf(ang) * sc;
          const float sn  = __sinf(ang) * sc;
          val = (col & 1) ? (val * cs + prt * sn) : (val * cs - prt * sn);
        }
        const int b = row >> 11, tl = row & (TPOS - 1);
        if (MODE == 3) {
          outf[(size_t)row * EDIM + col] = val;
        } else if (MODE == 2) {
          const int h = col >> 6, d = col & 63;
          outb[(((size_t)(b * NHEAD + h)) * HDIM + d) * TPOS + tl] = f2bf(val);
        } else {
          const int h = col >> 6, d = col & 63;
          outb[(((size_t)(b * NHEAD + h)) * TPOS + tl) * HDIM + d] = f2bf(val);
        }
      }
    }
  }
}

// ---------------------------------------------------------------------------
// Flash attention: block = 4 waves x 16 q-rows (64 rows) for one (b,h).
// S-tiles of 64; K/V staged to LDS by async DMA (double-buffered, shared by
// all 4 waves); online softmax; P -> LDS -> A-fragments for P@V.
// ---------------------------------------------------------------------------
__global__ __launch_bounds__(128) void attn_kernel(
    const unsigned short* __restrict__ Qb, const unsigned short* __restrict__ Kb,
    const unsigned short* __restrict__ Vt, unsigned short* __restrict__ Ob) {
  __shared__ __align__(16) unsigned short Klds[2][64][LDP];  // 18 KB
  __shared__ __align__(16) unsigned short Vlds[2][64][LDP];  // 18 KB
  __shared__ __align__(16) unsigned short Plds[4][16][LDP];  //  9 KB

  const int tid = threadIdx.x;
  const int lane = tid & 31, wave = tid >> 5;
  const int bh = blockIdx.y;
  const int q0 = blockIdx.x * 64 + wave * 16;

  const unsigned short* Qh = Qb + (size_t)bh * TPOS * HDIM;
  const unsigned short* Kh = Kb + (size_t)bh * TPOS * HDIM;   // [s][d]
  const unsigned short* Vh = Vt + (size_t)bh * HDIM * TPOS;   // [d][s]

  // LDS byte addresses for the async DMA destination (ISA: addr[31:0]).
  const unsigned kbase = (unsigned)(size_t)(void*)&Klds[0][0][0];
  const unsigned vbase = (unsigned)(size_t)(void*)&Vlds[0][0][0];
  const unsigned bufBytes = 64u * LDP * 2u;  // 9216 B per buffer

  // Stage one 64x64 K tile and one 64x64 V tile (8 KB each) for s0..s0+63.
  // 512 chunks of 16B per tile; 128 threads x 4 chunks; 8 async ops / thread.
  auto issue_tile = [&](int buf, int s0) {
#pragma unroll
    for (int j = 0; j < 4; ++j) {
      const int c = tid + j * 128;          // 0..511
      const int row = c >> 3;               // K: s_local ; V: d
      const int cc  = c & 7;                // 16B chunk within row
      const unsigned lrow = (unsigned)(row * LDP * 2 + cc * 16);
      async_ld16(kbase + buf * bufBytes + lrow,
                 (unsigned long long)(size_t)(const void*)
                     (Kh + (size_t)(s0 + row) * HDIM + cc * 8));
      async_ld16(vbase + buf * bufBytes + lrow,
                 (unsigned long long)(size_t)(const void*)
                     (Vh + (size_t)row * TPOS + s0 + cc * 8));
    }
  };

  v16bf aq[2];
  aq[0] = load_a_frag(Qh, HDIM, q0, 0, lane);
  aq[1] = load_a_frag(Qh, HDIM, q0, 32, lane);

  float mst[8], lst[8];
  v8f oacc[4];
#pragma unroll
  for (int r = 0; r < 8; ++r) { mst[r] = -3.0e38f; lst[r] = 0.f; }
#pragma unroll
  for (int dt = 0; dt < 4; ++dt) oacc[dt] = (v8f){0,0,0,0,0,0,0,0};

  issue_tile(0, 0);

  for (int it = 0; it < TPOS / 64; ++it) {
    const int buf = it & 1;
    if (it + 1 < TPOS / 64) {
      issue_tile(buf ^ 1, (it + 1) * 64);   // prefetch next tile
      // current tile's 8 ops are the oldest; async loads retire in order.
      asm volatile("s_wait_asynccnt 8" ::: "memory");
    } else {
      asm volatile("s_wait_asynccnt 0" ::: "memory");
    }
    __syncthreads();   // all waves' DMA chunks visible

    const unsigned short* Kt = &Klds[buf][0][0];
    const unsigned short* Vl = &Vlds[buf][0][0];

    // ---- scores: 16 x 64 per wave (K over d=64), B-frags from LDS ----
    v8f sacc[4];
#pragma unroll
    for (int nt = 0; nt < 4; ++nt) {
      sacc[nt] = (v8f){0,0,0,0,0,0,0,0};
#pragma unroll
      for (int kc = 0; kc < 2; ++kc) {
        v16bf kb = load_b_frag(Kt, LDP, nt * 16, kc * 32, lane);
        sacc[nt] = wmma_bf16(aq[kc], kb, sacc[nt]);
      }
    }
    // ---- online softmax (rows live in 16-lane halves; xor 1,2,4,8) ----
    float rmax[8], rsum[8], corr[8];
#pragma unroll
    for (int r = 0; r < 8; ++r) {
      float m = sacc[0][r];
      m = fmaxf(m, sacc[1][r]); m = fmaxf(m, sacc[2][r]); m = fmaxf(m, sacc[3][r]);
      rmax[r] = m;
    }
#pragma unroll
    for (int off = 1; off <= 8; off <<= 1)
#pragma unroll
      for (int r = 0; r < 8; ++r)
        rmax[r] = fmaxf(rmax[r], __shfl_xor(rmax[r], off, 32));
#pragma unroll
    for (int r = 0; r < 8; ++r) {
      const float mnew = fmaxf(mst[r], rmax[r]);
      corr[r] = __expf(mst[r] - mnew);
      mst[r] = mnew;
      float s = 0.f;
#pragma unroll
      for (int nt = 0; nt < 4; ++nt) {
        const float p = __expf(sacc[nt][r] - mnew);
        sacc[nt][r] = p;
        s += p;
      }
      rsum[r] = s;
    }
#pragma unroll
    for (int off = 1; off <= 8; off <<= 1)
#pragma unroll
      for (int r = 0; r < 8; ++r)
        rsum[r] += __shfl_xor(rsum[r], off, 32);
#pragma unroll
    for (int r = 0; r < 8; ++r) lst[r] = lst[r] * corr[r] + rsum[r];
#pragma unroll
    for (int dt = 0; dt < 4; ++dt)
#pragma unroll
      for (int r = 0; r < 8; ++r) oacc[dt][r] *= corr[r];

    // ---- P (C-layout) -> LDS row-major [row][s_local], as bf16 ----
#pragma unroll
    for (int nt = 0; nt < 4; ++nt)
#pragma unroll
      for (int r = 0; r < 8; ++r)
        Plds[wave][((lane >> 4) << 3) + r][nt * 16 + (lane & 15)] =
            f2bf(sacc[nt][r]);
    asm volatile("s_wait_dscnt 0" ::: "memory");  // cross-lane LDS RAW

    // ---- O += P @ V  (K over s_local; B-frags from LDS V tile) ----
#pragma unroll
    for (int kc = 0; kc < 2; ++kc) {
      v16bf pa = load_a_frag(&Plds[wave][0][0], LDP, 0, kc * 32, lane);
#pragma unroll
      for (int dt = 0; dt < 4; ++dt) {
        v16bf vb = load_b_frag(Vl, LDP, dt * 16, kc * 32, lane);
        oacc[dt] = wmma_bf16(pa, vb, oacc[dt]);
      }
    }
    __syncthreads();   // everyone done reading buf before it is overwritten
  }

  // ---- normalize and store to [b][t][e] bf16 ----
  float inv[8];
#pragma unroll
  for (int r = 0; r < 8; ++r) inv[r] = 1.0f / lst[r];
  const int b = bh >> 4, h = bh & 15;
#pragma unroll
  for (int dt = 0; dt < 4; ++dt)
#pragma unroll
    for (int r = 0; r < 8; ++r) {
      const int trow = q0 + ((lane >> 4) << 3) + r;
      const int e = h * HDIM + dt * 16 + (lane & 15);
      Ob[((size_t)b * TPOS + trow) * EDIM + e] = f2bf(oacc[dt][r] * inv[r]);
    }
}

// ---------------------------------------------------------------------------
extern "C" void kernel_launch(void* const* d_in, const int* in_sizes, int n_in,
                              void* d_out, int out_size, void* d_ws, size_t ws_size,
                              hipStream_t stream) {
  (void)in_sizes; (void)n_in; (void)out_size; (void)ws_size;
  const float* query = (const float*)d_in[0];
  const float* key   = (const float*)d_in[1];
  const float* value = (const float*)d_in[2];
  const float* wq = (const float*)d_in[3];  const float* bq = (const float*)d_in[4];
  const float* wk = (const float*)d_in[5];  const float* bk = (const float*)d_in[6];
  const float* wv = (const float*)d_in[7];  const float* bv = (const float*)d_in[8];
  const float* wo = (const float*)d_in[9];  const float* bo = (const float*)d_in[10];

  unsigned short* ws = (unsigned short*)d_ws;
  const size_t ACT = (size_t)NROWS * EDIM;   // 4M elems
  const size_t WSZ = (size_t)EDIM * EDIM;    // 1M elems
  unsigned short* Xq = ws;
  unsigned short* Xk = Xq + ACT;
  unsigned short* Xv = Xk + ACT;
  unsigned short* Wq = Xv + ACT;
  unsigned short* Wk = Wq + WSZ;
  unsigned short* Wv = Wk + WSZ;
  unsigned short* Wo = Wv + WSZ;
  unsigned short* Qb = Wo + WSZ;   // [B*H][T][D]
  unsigned short* Kb = Qb + ACT;   // [B*H][T][D]
  unsigned short* Vt = Kb + ACT;   // [B*H][D][T]
  unsigned short* At = Vt + ACT;   // [B*T][E]

  auto cv = [&](const float* s, unsigned short* d, size_t n) {
    int n4 = (int)(n / 4);
    conv_kernel<<<(n4 + 255) / 256, 256, 0, stream>>>(s, d, n4);
  };
  cv(query, Xq, ACT); cv(key, Xk, ACT); cv(value, Xv, ACT);
  cv(wq, Wq, WSZ); cv(wk, Wk, WSZ); cv(wv, Wv, WSZ); cv(wo, Wo, WSZ);

  const dim3 ggrid(EDIM / 64, NROWS / 128);  // (16, 32)
  gemm_kernel<0><<<ggrid, 256, 0, stream>>>(Xq, Wq, bq, Qb, nullptr);
  gemm_kernel<1><<<ggrid, 256, 0, stream>>>(Xk, Wk, bk, Kb, nullptr);
  gemm_kernel<2><<<ggrid, 256, 0, stream>>>(Xv, Wv, bv, Vt, nullptr);

  attn_kernel<<<dim3(TPOS / 64, 2 * NHEAD), 128, 0, stream>>>(Qb, Kb, Vt, At);

  gemm_kernel<3><<<ggrid, 256, 0, stream>>>(At, Wo, bo, nullptr, (float*)d_out);
}
